// FPBiencoderRankLearner_36919538876510
// MI455X (gfx1250) — matrix operand order repaired and checked by reading, not compiled, not run on hardware
//
#include <hip/hip_runtime.h>
#include <hip/hip_bf16.h>
#include <math.h>

typedef __attribute__((ext_vector_type(16))) _Float16 v16h;
typedef __attribute__((ext_vector_type(8)))  _Float16 v8h;
typedef __attribute__((ext_vector_type(4)))  _Float16 v4h;
typedef __attribute__((ext_vector_type(8)))  float    v8f;

#define NBITS   4096
#define DPROJ   512
#define BGRP    1024
#define TOTAL   32752
#define KSTEPS  (NBITS / 32)
#define LN_EPS  1e-5f
#define COS_EPS 1e-8f
#define INV_TEMP 10.0f

#define ASTRIDE 40          // LDS A row stride in halves (80B, 16B aligned)
#define HSTRIDE 512

// ---------------- wave32 reductions ----------------
__device__ __forceinline__ float wave_sum(float v) {
    #pragma unroll
    for (int o = 16; o > 0; o >>= 1) v += __shfl_xor(v, o, 32);
    return v;
}
__device__ __forceinline__ float wave_max(float v) {
    #pragma unroll
    for (int o = 16; o > 0; o >>= 1) v = fmaxf(v, __shfl_xor(v, o, 32));
    return v;
}
__device__ __forceinline__ int wave_min_i(int v) {
    #pragma unroll
    for (int o = 16; o > 0; o >>= 1) { int t = __shfl_xor(v, o, 32); v = (t < v) ? t : v; }
    return v;
}

__device__ __forceinline__ float gelu_exact(float x) {
    return 0.5f * x * (1.0f + erff(x * 0.70710678118654752440f));
}

// fast sigmoid: hardware v_exp_f32 + v_rcp_f32 (feeds f16 WMMA, approx is fine)
__device__ __forceinline__ float fast_sigmoid(float x) {
    return __builtin_amdgcn_rcpf(1.f + __expf(-x));
}

// ---------------- Wp -> WpT (f16, column-major [DPROJ][NBITS]) ----------------
__global__ void wpt_convert_kernel(const float* __restrict__ Wp, _Float16* __restrict__ WpT) {
    int idx = blockIdx.x * 256 + threadIdx.x;   // idx = n*NBITS + k
    int k = idx & (NBITS - 1);
    int n = idx >> 12;
    WpT[idx] = (_Float16)Wp[(size_t)k * DPROJ + n];
}

// ---------------- exclusive scan of counts (single 1024-thread block) ----------------
__global__ void scan_kernel(const int* __restrict__ counts, int* __restrict__ starts) {
    __shared__ int s[BGRP];
    int t = threadIdx.x;
    int c = counts[t];
    s[t] = c;
    __syncthreads();
    for (int off = 1; off < BGRP; off <<= 1) {
        int v = (t >= off) ? s[t - off] : 0;
        __syncthreads();
        s[t] += v;
        __syncthreads();
    }
    starts[t] = s[t] - c;
}

__global__ void segfill_kernel(const int* __restrict__ counts, const int* __restrict__ starts,
                               int* __restrict__ seg) {
    int b = blockIdx.x;
    int st = starts[b], cnt = counts[b];
    for (int i = threadIdx.x; i < cnt; i += blockDim.x) seg[st + i] = b;
}

// ---------------- A tile loader: f32 global -> f16 LDS (optional sigmoid) ----------------
// Tile is MT x 32 (MT = MTILES*16); 256 threads, each converts MT*32/256 floats.
template <bool SIG, int MTILES>
__device__ __forceinline__ void load_a_tile(_Float16* dst, const float* __restrict__ A,
                                            int mBase, int Mrows, int kbase) {
    constexpr int MT  = MTILES * 16;
    constexpr int TPR = 256 / MT;           // threads per row (8 or 4)
    constexpr int FPT = 32 / TPR;           // floats per thread (4 or 8)
    const int t   = threadIdx.x;
    const int row = t / TPR;
    const int cb  = (t % TPR) * FPT;
    const int g   = mBase + row;
    #pragma unroll
    for (int c = 0; c < FPT; c += 4) {
        float4 v;
        if (g < Mrows) {
            v = *(const float4*)(A + (size_t)g * NBITS + kbase + cb + c);
        } else {
            v = make_float4(0.f, 0.f, 0.f, 0.f);
        }
        if (SIG) {
            v.x = fast_sigmoid(v.x);
            v.y = fast_sigmoid(v.y);
            v.z = fast_sigmoid(v.z);
            v.w = fast_sigmoid(v.w);
        }
        v4h h;
        h[0] = (_Float16)v.x; h[1] = (_Float16)v.y;
        h[2] = (_Float16)v.z; h[3] = (_Float16)v.w;
        *(v4h*)(dst + row * ASTRIDE + cb + c) = h;
    }
}

// ---------------- fused projector GEMM (+GELU+LN) with two epilogues ----------------
// WRITE_P: write P rows + norms.   !WRITE_P: fuse cosine score vs P[seg] -> xs.
template <bool SIG, bool WRITE_P, int MTILES>
__global__ __launch_bounds__(256)
void gemm_proj_kernel(const float* __restrict__ Asrc, const _Float16* __restrict__ WpT,
                      const float* __restrict__ bp, const float* __restrict__ gamma,
                      const float* __restrict__ beta, int Mrows,
                      float* __restrict__ Pout, float* __restrict__ pnormOut,
                      const int* __restrict__ seg, const float* __restrict__ Pin,
                      const float* __restrict__ pnormIn, float* __restrict__ xs) {
    constexpr int MT = MTILES * 16;
    __shared__ __align__(16) _Float16 Alds[2][MT * ASTRIDE];
    __shared__ float H[MT * HSTRIDE];

    const int w     = threadIdx.x >> 5;
    const int lane  = threadIdx.x & 31;
    const int hi    = lane >> 4;
    const int ln15  = lane & 15;
    const int mBase = blockIdx.x * MT;

    v8f acc[MTILES][4];
    #pragma unroll
    for (int i = 0; i < MTILES; ++i)
        #pragma unroll
        for (int j = 0; j < 4; ++j) acc[i][j] = v8f{};

    union AU { v16h v; v8h h[2]; };

    // B fragment pointer base: per-lane row of WpT (N = lane&15), K-contiguous.
    // B layout: element e -> K = 16*hi + e, N = lane&15.
    const _Float16* bbase[4];
    #pragma unroll
    for (int nt = 0; nt < 4; ++nt)
        bbase[nt] = WpT + (size_t)(w * 64 + nt * 16 + ln15) * NBITS + hi * 16;

    // One k-step: stage next A tile, load A frags (cur LDS buf), prefetch next
    // k-step's B frags into 'bn' registers, then run WMMAs on current 'bc'.
    auto kstep = [&](int kb, int cur, v16h* bc, v16h* bn) {
        __syncthreads();
        if (kb + 1 < KSTEPS)
            load_a_tile<SIG, MTILES>(Alds[cur ^ 1], Asrc, mBase, Mrows, (kb + 1) * 32);
        if (kb + 2 < KSTEPS) {               // L2 prefetch of A stream (global_prefetch_b8)
            const int t = threadIdx.x, row = t / (256 / MT), g = mBase + row;
            if (g < Mrows)
                __builtin_prefetch(Asrc + (size_t)g * NBITS + (kb + 2) * 32 +
                                   (t % (256 / MT)) * (32 / (256 / MT)), 0, 1);
        }
        // A layout: element e -> K = 16*hi + e, M = lane&15.
        AU afrag[MTILES];
        #pragma unroll
        for (int mt = 0; mt < MTILES; ++mt) {
            const _Float16* base = &Alds[cur][(mt * 16 + ln15) * ASTRIDE + hi * 16];
            afrag[mt].h[0] = *(const v8h*)(base);
            afrag[mt].h[1] = *(const v8h*)(base + 8);
        }
        if (kb + 1 < KSTEPS) {
            #pragma unroll
            for (int nt = 0; nt < 4; ++nt)
                bn[nt] = *(const v16h*)(bbase[nt] + (kb + 1) * 32);
        }
        #pragma unroll
        for (int nt = 0; nt < 4; ++nt)
            #pragma unroll
            for (int mt = 0; mt < MTILES; ++mt)
                acc[mt][nt] = __builtin_amdgcn_wmma_f32_16x16x32_f16(
                    false, afrag[mt].v, false, bc[nt], (short)0, acc[mt][nt], false, false);
    };

    v16h bufA[4], bufB[4];
    #pragma unroll
    for (int nt = 0; nt < 4; ++nt) bufA[nt] = *(const v16h*)(bbase[nt]);
    load_a_tile<SIG, MTILES>(Alds[0], Asrc, mBase, Mrows, 0);

    for (int kb = 0; kb < KSTEPS; kb += 2) {   // unrolled x2: register double-buffer for B
        kstep(kb,     0, bufA, bufB);
        kstep(kb + 1, 1, bufB, bufA);
    }

    // Writeback with bias + exact GELU.  D layout: VGPR r -> M = r + 8*hi, N = lane&15.
    #pragma unroll
    for (int mt = 0; mt < MTILES; ++mt)
        #pragma unroll
        for (int nt = 0; nt < 4; ++nt) {
            const int col  = w * 64 + nt * 16 + ln15;
            const float bb = bp[col];
            #pragma unroll
            for (int r = 0; r < 8; ++r) {
                const int mloc = mt * 16 + hi * 8 + r;
                H[mloc * HSTRIDE + col] = gelu_exact(acc[mt][nt][r] + bb);
            }
        }
    __syncthreads();

    // LayerNorm + epilogue; each wave owns MT/8 rows.
    for (int q = 0; q < MT / 8; ++q) {
        const int rl   = w * (MT / 8) + q;
        const int rowg = mBase + rl;
        float gv[16];
        float s1 = 0.f, s2 = 0.f;
        #pragma unroll
        for (int j = 0; j < 16; ++j) {
            float g = H[rl * HSTRIDE + lane + 32 * j];
            gv[j] = g; s1 += g; s2 += g * g;
        }
        s1 = wave_sum(s1);
        s2 = wave_sum(s2);
        const float mu  = s1 * (1.f / (float)DPROJ);
        const float var = s2 * (1.f / (float)DPROJ) - mu * mu;
        const float inv = rsqrtf(var + LN_EPS);

        if (WRITE_P) {
            float ny = 0.f;
            #pragma unroll
            for (int j = 0; j < 16; ++j) {
                const int col = lane + 32 * j;
                float y = (gv[j] - mu) * inv * gamma[col] + beta[col];
                if (rowg < Mrows) Pout[(size_t)rowg * DPROJ + col] = y;
                ny += y * y;
            }
            ny = wave_sum(ny);
            if (lane == 0 && rowg < Mrows) pnormOut[rowg] = sqrtf(ny);
        } else {
            const int sg = (rowg < Mrows) ? seg[rowg] : 0;
            float ny = 0.f, pd = 0.f;
            #pragma unroll
            for (int j = 0; j < 16; ++j) {
                const int col = lane + 32 * j;
                float y  = (gv[j] - mu) * inv * gamma[col] + beta[col];
                float pv = Pin[(size_t)sg * DPROJ + col];
                ny += y * y;
                pd += y * pv;
            }
            ny = wave_sum(ny);
            pd = wave_sum(pd);
            if (lane == 0 && rowg < Mrows) {
                float denom = fmaxf(sqrtf(ny) * pnormIn[sg], COS_EPS);
                xs[rowg] = (pd / denom) * INV_TEMP;
            }
        }
    }
}

// ---------------- per-group listwise CE (one wave per group) ----------------
__global__ void group_loss_kernel(const float* __restrict__ xs, const int* __restrict__ starts,
                                  const int* __restrict__ counts,
                                  const unsigned char* __restrict__ labels,
                                  float* __restrict__ gloss) {
    const int b = blockIdx.x, lane = threadIdx.x;
    const int st = starts[b], cnt = counts[b];

    int fp = 0x7fffffff;
    for (int i = lane; i < cnt; i += 32)
        if (labels[st + i]) fp = (i < fp) ? i : fp;
    fp = wave_min_i(fp);

    float m = -INFINITY;
    for (int i = lane; i < cnt; i += 32) {
        bool lab = labels[st + i] != 0;
        if (!lab || i == fp) m = fmaxf(m, xs[st + i]);
    }
    m = wave_max(m);

    float se = 0.f;
    for (int i = lane; i < cnt; i += 32) {
        bool lab = labels[st + i] != 0;
        if (!lab || i == fp) se += expf(xs[st + i] - m);
    }
    se = wave_sum(se);

    if (lane == 0) {
        float spos = (fp < cnt) ? xs[st + fp] : 0.f;
        gloss[b] = m + logf(se) - spos;
    }
}

__global__ void finalize_kernel(const float* __restrict__ gloss, float* __restrict__ out) {
    __shared__ float red[256];
    float s = 0.f;
    for (int i = threadIdx.x; i < BGRP; i += 256) s += gloss[i];
    red[threadIdx.x] = s;
    __syncthreads();
    for (int o = 128; o > 0; o >>= 1) {
        if (threadIdx.x < o) red[threadIdx.x] += red[threadIdx.x + o];
        __syncthreads();
    }
    if (threadIdx.x == 0) out[0] = red[0] * (1.f / (float)BGRP);
}

// ---------------- host launch ----------------
extern "C" void kernel_launch(void* const* d_in, const int* in_sizes, int n_in,
                              void* d_out, int out_size, void* d_ws, size_t ws_size,
                              hipStream_t stream) {
    (void)in_sizes; (void)n_in; (void)out_size; (void)ws_size;
    const float*         logits  = (const float*)d_in[0];
    /* d_in[1] = true_fp : unused by the reference */
    const float*         cand_fp = (const float*)d_in[2];
    const int*           counts  = (const int*)d_in[3];
    const unsigned char* labels  = (const unsigned char*)d_in[4];
    const float*         Wp      = (const float*)d_in[5];
    const float*         bp      = (const float*)d_in[6];
    const float*         gamma   = (const float*)d_in[7];
    const float*         beta    = (const float*)d_in[8];
    float* out = (float*)d_out;

    char* ws = (char*)d_ws;                       // ~6.6 MB total
    _Float16* WpT   = (_Float16*)(ws + 0);        // 4 MB   f16 [512][4096]
    float*    P     = (float*)(ws + 4194304);     // 2 MB   f32 [1024][512]
    float*    pnorm = (float*)(ws + 6291456);     // 4 KB
    int*      starts= (int*)(ws + 6295552);       // 4 KB
    int*      seg   = (int*)(ws + 6299648);       // 128 KB
    float*    xsc   = (float*)(ws + 6430720);     // 128 KB
    float*    gloss = (float*)(ws + 6561792);     // 4 KB

    wpt_convert_kernel<<<(NBITS * DPROJ) / 256, 256, 0, stream>>>(Wp, WpT);
    scan_kernel<<<1, BGRP, 0, stream>>>(counts, starts);
    segfill_kernel<<<BGRP, 64, 0, stream>>>(counts, starts, seg);

    // GEMM-P: P = LN(GELU(sigmoid(logits) @ Wp + bp)) and row norms.  M-tile 32 -> 32 blocks.
    gemm_proj_kernel<true, true, 2><<<BGRP / 32, 256, 0, stream>>>(
        logits, WpT, bp, gamma, beta, BGRP, P, pnorm, nullptr, nullptr, nullptr, nullptr);

    // GEMM-C fused with cosine score vs P[seg].  M-tile 64 -> 512 blocks, halves WpT L2 traffic.
    gemm_proj_kernel<false, false, 4><<<(TOTAL + 63) / 64, 256, 0, stream>>>(
        cand_fp, WpT, bp, gamma, beta, TOTAL, nullptr, nullptr, seg, P, pnorm, xsc);

    group_loss_kernel<<<BGRP, 32, 0, stream>>>(xsc, starts, counts, labels, gloss);
    finalize_kernel<<<1, 256, 0, stream>>>(gloss, out);
}
